// VarMaskedRNNBase_58213986730063
// MI455X (gfx1250) — matrix operand order repaired and checked by reading, not compiled, not run on hardware
//
#include <hip/hip_runtime.h>
#include <hip/hip_bf16.h>
#include <math.h>

// ---------------------------------------------------------------------------
// VarMaskedRNN (2-layer, masked) for MI455X / gfx1250, wave32 + WMMA bf16.
//   layer: pre = X @ W_ih^T + b_ih  (big WMMA GEMM, M=65536,K=N=1024)
//          scan: h = blend(tanh(pre_t + h @ W_hh^T + b_hh), h, mask_t)
//   scan is a persistent 64-block kernel with a grid-wide atomic barrier
//   per timestep; W_hh fragments are register-resident across all T steps and
//   all 16 h fragments for a step are loaded (sched-fenced) before any WMMA.
// ---------------------------------------------------------------------------

typedef __attribute__((ext_vector_type(16))) __bf16 v16bf;
typedef __attribute__((ext_vector_type(8)))  __bf16 v8bf;
typedef __attribute__((ext_vector_type(8)))  float  v8f;

__device__ __forceinline__ v16bf concat8(v8bf lo, v8bf hi) {
  return __builtin_shufflevector(lo, hi, 0,1,2,3,4,5,6,7,8,9,10,11,12,13,14,15);
}

// A fragment 16x32 (MxK), bf16 row-major source, leading dim lda.
// ISA layout: lanes 0-15: M=lane, K={0..7,16..23}; lanes 16-31: M=lane-16, K={8..15,24..31}
template<bool PREFETCH>
__device__ __forceinline__ v16bf load_a_bf16(const __bf16* A, int lda, int m0, int k0, int lane) {
  const int row  = m0 + (lane & 15);
  const int koff = (lane < 16) ? 0 : 8;
  const __bf16* p = A + (size_t)row * lda + (size_t)(k0 + koff);
  if constexpr (PREFETCH) __builtin_prefetch(p + 64, 0, 3);   // next K-block
  v8bf lo = *reinterpret_cast<const v8bf*>(p);
  v8bf hi = *reinterpret_cast<const v8bf*>(p + 16);
  return concat8(lo, hi);
}

// Same fragment but fp32 source, converted to bf16 on the fly.
__device__ __forceinline__ v16bf load_a_f32(const float* A, int lda, int m0, int k0, int lane) {
  const int row  = m0 + (lane & 15);
  const int koff = (lane < 16) ? 0 : 8;
  const float* p = A + (size_t)row * lda + (size_t)(k0 + koff);
  __builtin_prefetch(p + 64, 0, 3);
  float t[16];
  *reinterpret_cast<float4*>(t + 0)  = *reinterpret_cast<const float4*>(p + 0);
  *reinterpret_cast<float4*>(t + 4)  = *reinterpret_cast<const float4*>(p + 4);
  *reinterpret_cast<float4*>(t + 8)  = *reinterpret_cast<const float4*>(p + 16);
  *reinterpret_cast<float4*>(t + 12) = *reinterpret_cast<const float4*>(p + 20);
  v16bf a;
  #pragma unroll
  for (int i = 0; i < 16; ++i) a[i] = (__bf16)t[i];
  return a;
}

// B fragment 32x16 (KxN) with B[k][n] = W[n][k], W row-major [N][K] bf16.
// ISA layout: column N = n0 + (lane&15); lanes 0-15 K={0..7,16..23}, lanes 16-31 K={8..15,24..31}
__device__ __forceinline__ v16bf load_b_w(const __bf16* W, int ldw, int n0, int k0, int lane) {
  const int n    = n0 + (lane & 15);
  const int koff = (lane < 16) ? 0 : 8;
  const __bf16* p = W + (size_t)n * ldw + (size_t)(k0 + koff);
  v8bf lo = *reinterpret_cast<const v8bf*>(p);
  v8bf hi = *reinterpret_cast<const v8bf*>(p + 16);
  return concat8(lo, hi);
}

// ---------------------------------------------------------------------------
// GEMM: C[m][n] = sum_k A[m][k] * W[n][k] + bias[n].  One wave -> 16x64 strip.
// Throughput-bound; massive grid hides memory latency.
// ---------------------------------------------------------------------------
template<bool A_F32>
__global__ __launch_bounds__(256)
void wmma_gemm_bias(const void* __restrict__ Av, const __bf16* __restrict__ W,
                    const float* __restrict__ bias, float* __restrict__ C,
                    int M, int K, int N) {
  const int lane   = threadIdx.x & 31;
  const int wave   = threadIdx.x >> 5;
  const int waveId = blockIdx.x * 8 + wave;
  const int ngrps  = N >> 6;                // 64-column groups
  const int m0     = (waveId / ngrps) * 16;
  const int n0     = (waveId % ngrps) * 64;
  if (m0 >= M) return;

  v8f acc[4] = {};
  for (int k0 = 0; k0 < K; k0 += 32) {
    v16bf a;
    if constexpr (A_F32) a = load_a_f32((const float*)Av, K, m0, k0, lane);
    else                 a = load_a_bf16<true>((const __bf16*)Av, K, m0, k0, lane);
    #pragma unroll
    for (int nt = 0; nt < 4; ++nt) {
      v16bf b = load_b_w(W, K, n0 + nt * 16, k0, lane);
      acc[nt] = __builtin_amdgcn_wmma_f32_16x16x32_bf16(
          false, a, false, b, (short)0, acc[nt], false, false);
    }
  }

  // C/D layout: VGPR r, lane l -> M = m0 + r + 8*(l>>4), N = n0 + (l&15)
  // pre is a 256MB one-shot stream: bypass caches with non-temporal stores.
  const int rbase = m0 + ((lane >> 4) << 3);
  const int col0  = n0 + (lane & 15);
  #pragma unroll
  for (int nt = 0; nt < 4; ++nt) {
    const int c  = col0 + nt * 16;
    const float bb = bias[c];
    #pragma unroll
    for (int r = 0; r < 8; ++r)
      __builtin_nontemporal_store(acc[nt][r] + bb, &C[(size_t)(rbase + r) * N + c]);
  }
}

// ---------------------------------------------------------------------------
// Persistent masked-RNN scan. 64 blocks x 256 threads; block b owns columns
// [b*16, b*16+16); 8 waves split K=1024 into 128-wide slices; LDS reduction.
// ---------------------------------------------------------------------------
__device__ __forceinline__ void grid_barrier(unsigned* cnt, unsigned* gen, int nblocks) {
  __threadfence();
  __syncthreads();
  if (threadIdx.x == 0) {
    unsigned g = __hip_atomic_load(gen, __ATOMIC_ACQUIRE, __HIP_MEMORY_SCOPE_AGENT);
    unsigned a = __hip_atomic_fetch_add(cnt, 1u, __ATOMIC_ACQ_REL, __HIP_MEMORY_SCOPE_AGENT);
    if (a == (unsigned)nblocks - 1u) {
      __hip_atomic_store(cnt, 0u, __ATOMIC_RELAXED, __HIP_MEMORY_SCOPE_AGENT);
      __hip_atomic_fetch_add(gen, 1u, __ATOMIC_RELEASE, __HIP_MEMORY_SCOPE_AGENT);
    } else {
      while (__hip_atomic_load(gen, __ATOMIC_ACQUIRE, __HIP_MEMORY_SCOPE_AGENT) == g)
        __builtin_amdgcn_s_sleep(1);
    }
  }
  __syncthreads();
}

__global__ __launch_bounds__(256)
__attribute__((amdgpu_waves_per_eu(1)))     // only 2 waves/SIMD needed resident
void scan_layer(const float* __restrict__ pre,      // [T,B,H]
                const float* __restrict__ mask,     // [T,B]
                const __bf16* __restrict__ Whh,     // [H,H] bf16
                const float* __restrict__ bhh,      // [H]
                float* __restrict__ out_f32,        // [T,B,H] or null
                __bf16* __restrict__ out_bf,        // [T,B,H] or null
                float* __restrict__ hT_out,         // [B,H]
                float* __restrict__ hbuf,           // [2,B,H] f32 ping-pong
                __bf16* __restrict__ hbf,           // [2,B,H] bf16 ping-pong
                unsigned* __restrict__ bar_cnt, unsigned* __restrict__ bar_gen,
                int T, int nblocks) {
  constexpr int B = 64, H = 1024;
  __shared__ float red[8][64][16];                  // 32 KB
  const int lane  = threadIdx.x & 31;
  const int wave  = threadIdx.x >> 5;
  const int n0    = blockIdx.x * 16;
  const int kbase = wave * 128;

  // Hoist this wave's W_hh fragments (16 cols x 128 K) into registers: 32 VGPRs.
  v16bf bfr[4];
  #pragma unroll
  for (int kk = 0; kk < 4; ++kk)
    bfr[kk] = load_b_w(Whh, H, n0, kbase + kk * 32, lane);
  __builtin_amdgcn_sched_barrier(0);   // keep the hoist out of the T-loop

  for (int t = 0; t < T; ++t) {
    if (t > 0) {
      const __bf16* hsrc = hbf + (size_t)((t + 1) & 1) * B * H;

      // Issue ALL 32 b128 loads for this step, then fence the scheduler so
      // they cannot be sunk below the WMMAs: one L2 round-trip amortized over
      // 16 WMMAs instead of 16 serialized load->wait->wmma chains.
      v16bf af[16];
      #pragma unroll
      for (int kk = 0; kk < 4; ++kk)
        #pragma unroll
        for (int mt = 0; mt < 4; ++mt)
          af[kk * 4 + mt] = load_a_bf16<false>(hsrc, H, mt * 16, kbase + kk * 32, lane);
      __builtin_amdgcn_sched_barrier(0);

      v8f acc[4] = {};
      #pragma unroll
      for (int kk = 0; kk < 4; ++kk)
        #pragma unroll
        for (int mt = 0; mt < 4; ++mt)
          acc[mt] = __builtin_amdgcn_wmma_f32_16x16x32_bf16(
              false, af[kk * 4 + mt], false, bfr[kk], (short)0, acc[mt], false, false);

      const int rb = (lane >> 4) << 3;
      const int c  = lane & 15;
      #pragma unroll
      for (int mt = 0; mt < 4; ++mt)
        #pragma unroll
        for (int r = 0; r < 8; ++r)
          red[wave][mt * 16 + rb + r][c] = acc[mt][r];
    }
    __syncthreads();

    // reduce 8 K-slices + pointwise: 64x16 elements, 4 per thread
    float* hw        = hbuf + (size_t)(t & 1) * B * H;
    __bf16* hbw      = hbf  + (size_t)(t & 1) * B * H;
    const float* hr  = hbuf + (size_t)((t + 1) & 1) * B * H;
    #pragma unroll
    for (int i = 0; i < 4; ++i) {
      const int idx = threadIdx.x + i * 256;
      const int m   = idx >> 4;
      const int n   = idx & 15;
      const int col = n0 + n;
      float s = 0.f;
      if (t > 0) {
        #pragma unroll
        for (int w2 = 0; w2 < 8; ++w2) s += red[w2][m][n];
      }
      // pre is a one-shot stream -> non-temporal load
      const float p     = __builtin_nontemporal_load(&pre[(size_t)t * B * H + (size_t)m * H + col]) + bhh[col];
      const float hprev = (t > 0) ? hr[(size_t)m * H + col] : 0.f;
      const float mval  = mask[t * B + m];
      const float hcand = tanhf(s + p);
      const float hnew  = hprev + (hcand - hprev) * mval;
      hw[(size_t)m * H + col]  = hnew;                 // hot: keep cached
      hbw[(size_t)m * H + col] = (__bf16)hnew;         // hot: keep cached
      if (out_f32) __builtin_nontemporal_store(hnew, &out_f32[(size_t)t * B * H + (size_t)m * H + col]);
      if (out_bf)  __builtin_nontemporal_store((__bf16)hnew, &out_bf[(size_t)t * B * H + (size_t)m * H + col]);
      if (t == T - 1) hT_out[(size_t)m * H + col] = hnew;
    }
    grid_barrier(bar_cnt, bar_gen, nblocks);
  }
}

// ---------------------------------------------------------------------------
__global__ void cvt_f32_to_bf16(const float* __restrict__ src, __bf16* __restrict__ dst, int n) {
  int i = blockIdx.x * 256 + threadIdx.x;
  if (i < n) dst[i] = (__bf16)src[i];
}

__global__ void init_barrier(unsigned* p) {
  if (threadIdx.x == 0) { p[0] = 0u; p[1] = 0u; }
}

// ---------------------------------------------------------------------------
extern "C" void kernel_launch(void* const* d_in, const int* in_sizes, int n_in,
                              void* d_out, int out_size, void* d_ws, size_t ws_size,
                              hipStream_t stream) {
  (void)in_sizes; (void)n_in; (void)out_size; (void)ws_size;
  constexpr int T = 1024, B = 64, D = 1024, H = 1024;

  const float* x    = (const float*)d_in[0];
  const float* mask = (const float*)d_in[1];
  const float* Wih0 = (const float*)d_in[2];
  const float* Whh0 = (const float*)d_in[3];
  const float* bih0 = (const float*)d_in[4];
  const float* bhh0 = (const float*)d_in[5];
  const float* Wih1 = (const float*)d_in[6];
  const float* Whh1 = (const float*)d_in[7];
  const float* bih1 = (const float*)d_in[8];
  const float* bhh1 = (const float*)d_in[9];
  float* out = (float*)d_out;

  // workspace carve-up (256B aligned)
  char* ws = (char*)d_ws;
  size_t off = 0;
  auto carve = [&](size_t bytes) -> void* {
    void* p = ws + off;
    off = (off + bytes + 255) & ~(size_t)255;
    return p;
  };
  float*  pre    = (float*) carve((size_t)T * B * H * 4);   // 256 MB, reused per layer
  __bf16* out0bf = (__bf16*)carve((size_t)T * B * H * 2);   // 128 MB
  __bf16* Wih0b  = (__bf16*)carve((size_t)H * D * 2);
  __bf16* Whh0b  = (__bf16*)carve((size_t)H * H * 2);
  __bf16* Wih1b  = (__bf16*)carve((size_t)H * H * 2);
  __bf16* Whh1b  = (__bf16*)carve((size_t)H * H * 2);
  float*  hbuf   = (float*) carve((size_t)2 * B * H * 4);
  __bf16* hbf    = (__bf16*)carve((size_t)2 * B * H * 2);
  unsigned* bar  = (unsigned*)carve(256);

  // 1) weights -> bf16, barrier init
  const int nw = H * D;
  cvt_f32_to_bf16<<<(nw + 255) / 256, 256, 0, stream>>>(Wih0, Wih0b, nw);
  cvt_f32_to_bf16<<<(nw + 255) / 256, 256, 0, stream>>>(Whh0, Whh0b, nw);
  cvt_f32_to_bf16<<<(nw + 255) / 256, 256, 0, stream>>>(Wih1, Wih1b, nw);
  cvt_f32_to_bf16<<<(nw + 255) / 256, 256, 0, stream>>>(Whh1, Whh1b, nw);
  init_barrier<<<1, 32, 0, stream>>>(bar);

  const int M = T * B;
  const dim3 gemm_grid((unsigned)((M / 16) * (H / 64) / 8));  // 8 waves/block

  // 2) layer 0: pre = x @ W_ih0^T + b_ih0  (fp32 A, converted on the fly)
  wmma_gemm_bias<true><<<gemm_grid, 256, 0, stream>>>(x, Wih0b, bih0, pre, M, D, H);

  // 3) layer 0 scan -> out0 (bf16 for next GEMM), hn[0]
  scan_layer<<<64, 256, 0, stream>>>(pre, mask, Whh0b, bhh0,
                                     nullptr, out0bf,
                                     out + (size_t)T * B * H,
                                     hbuf, hbf, bar, bar + 1, T, 64);

  // 4) layer 1: pre = out0 @ W_ih1^T + b_ih1 (bf16 A)
  wmma_gemm_bias<false><<<gemm_grid, 256, 0, stream>>>(out0bf, Wih1b, bih1, pre, M, H, H);

  // 5) layer 1 scan -> d_out[0:T*B*H] (fp32), hn[1]
  scan_layer<<<64, 256, 0, stream>>>(pre, mask, Whh1b, bhh1,
                                     out, nullptr,
                                     out + (size_t)T * B * H + (size_t)B * H,
                                     hbuf, hbf, bar, bar + 1, T, 64);
}
